// ConvLSTMModel_27204322853090
// MI455X (gfx1250) — matrix operand order repaired and checked by reading, not compiled, not run on hardware
//
#include <hip/hip_runtime.h>

// Reference reduces to a 2-layer LSTM (center tap only of the 3x3 convs).
// All GEMMs run through V_WMMA_F32_16X16X32_F16, fp32 accum, fp32 cell state.
// v2: software-pipelined inner loop (double-buffered A/B fragments) so next
// chunk's L2 loads overlap the current chunk's 4 WMMAs (run is latency-bound).

#define HD   512       // HIDDEN_DIM
#define BB   128       // BATCH
#define TT   256       // SEQ
#define IND  256       // INPUT_DIM
#define NC   1000      // NUM_CLASSES
#define NCP  1008      // NC padded to multiple of 16

typedef _Float16 v16h __attribute__((ext_vector_type(16)));
typedef _Float16 v8h  __attribute__((ext_vector_type(8)));
typedef float    v8f  __attribute__((ext_vector_type(8)));

// ---- WMMA fragment helpers (layouts per CDNA5 ISA §7.12.2) ----

// A (16x32 f16, MxK): lane L holds row m=L&15; half=L>>4 selects K-groups.
// elements 0..7 -> K = 8*half + e ; elements 8..15 -> K = 16 + 8*half + (e-8)
static __device__ inline v16h load_a_frag(const _Float16* __restrict__ rowp, int half) {
  const v8h lo = *(const v8h*)(rowp + 8 * half);        // 16B -> global_load_b128
  const v8h hi = *(const v8h*)(rowp + 16 + 8 * half);   // 16B -> global_load_b128
  return __builtin_shufflevector(lo, hi, 0,1,2,3,4,5,6,7,8,9,10,11,12,13,14,15);
}

// B (32x16 f16, KxN): lane L holds column n=L&15; elements 0..7 -> K = 8*half + e ;
// elements 8..15 -> K = 16 + 8*half + (e-8). Weight stored row-major [N][K].
static __device__ inline v16h load_b_frag(const _Float16* __restrict__ wrow, int kk, int half) {
  const v8h lo = *(const v8h*)(wrow + kk + 8 * half);        // 16B -> global_load_b128
  const v8h hi = *(const v8h*)(wrow + kk + 16 + 8 * half);   // 16B -> global_load_b128
  return __builtin_shufflevector(lo, hi, 0,1,2,3,4,5,6,7,8,9,10,11,12,13,14,15);
}

static __device__ inline v8f wmma_f16(v16h a, v16h b, v8f c) {
  return __builtin_amdgcn_wmma_f32_16x16x32_f16(false, a, false, b, (short)0, c, false, false);
}

static __device__ inline float sigmoid_fast(float x) { return 1.0f / (1.0f + __expf(-x)); }
static __device__ inline float tanh_fast(float x)    { return 1.0f - 2.0f / (__expf(2.0f * x) + 1.0f); }

// ---- One LSTM cell step: gates = [Ax | Ah] @ W^T + bias; update c, emit h ----
// A K-region: k <  Kx  -> Ax with row stride Sx
//             k >= Kx  -> Ah with row stride HD
// Each wave: one 16(batch) x 16(hidden) tile, 4 gate accumulators.
__global__ __launch_bounds__(128) void lstm_cell_wmma(
    const _Float16* __restrict__ Ax, int Sx, int Kx,
    const _Float16* __restrict__ Ah,
    const _Float16* __restrict__ W, int Ktot,
    const float* __restrict__ bias,
    float* __restrict__ Cst,
    _Float16* __restrict__ Hout)
{
  const int lane = threadIdx.x & 31;
  const int gw   = blockIdx.x * 4 + (threadIdx.x >> 5);   // 256 waves total
  const int m0   = (gw >> 5) * 16;                        // batch tile   (8)
  const int n0   = (gw & 31) * 16;                        // hidden tile  (32)
  const int half = lane >> 4;
  const int nl   = lane & 15;

  const _Float16* w_i = W + (size_t)(n0 + nl) * Ktot;     // gate rows: i,f,o,g
  const _Float16* w_f = w_i + (size_t)(1 * HD) * Ktot;
  const _Float16* w_o = w_i + (size_t)(2 * HD) * Ktot;
  const _Float16* w_g = w_i + (size_t)(3 * HD) * Ktot;

  const _Float16* ax_row = Ax + (size_t)(m0 + nl) * Sx;
  const _Float16* ah_row = Ah + (size_t)(m0 + nl) * HD;

  v8f acc_i = {}, acc_f = {}, acc_o = {}, acc_g = {};

  // --- software pipeline: fragments for chunk kk held while chunk kk+32 loads ---
  v16h a  = load_a_frag(ax_row, half);                    // kk = 0 (always x region)
  v16h bi = load_b_frag(w_i, 0, half);
  v16h bf = load_b_frag(w_f, 0, half);
  v16h bo = load_b_frag(w_o, 0, half);
  v16h bg = load_b_frag(w_g, 0, half);

  for (int kk = 0; kk < Ktot; kk += 32) {
    // prefetch next chunk (last iteration re-fetches chunk 0; result unused)
    const int kn = (kk + 32 < Ktot) ? (kk + 32) : 0;      // scalar-uniform
    const _Float16* arow_n = (kn < Kx) ? (ax_row + kn) : (ah_row + (kn - Kx));
    const v16h a2  = load_a_frag(arow_n, half);
    const v16h bi2 = load_b_frag(w_i, kn, half);
    const v16h bf2 = load_b_frag(w_f, kn, half);
    const v16h bo2 = load_b_frag(w_o, kn, half);
    const v16h bg2 = load_b_frag(w_g, kn, half);

    // consume current chunk while next chunk's loads are in flight
    acc_i = wmma_f16(a, bi, acc_i);
    acc_f = wmma_f16(a, bf, acc_f);
    acc_o = wmma_f16(a, bo, acc_o);
    acc_g = wmma_f16(a, bg, acc_g);

    a = a2; bi = bi2; bf = bf2; bo = bo2; bg = bg2;
  }

  const int   n  = n0 + nl;
  const float bI = bias[n];
  const float bF = bias[HD + n];
  const float bO = bias[2 * HD + n];
  const float bG = bias[3 * HD + n];

  // C/D layout: vgpr r, lane L -> M = r + 8*half, N = lane&15
  #pragma unroll
  for (int r = 0; r < 8; ++r) {
    const int    b   = m0 + r + 8 * half;
    const size_t idx = (size_t)b * HD + n;
    const float iv = sigmoid_fast(acc_i[r] + bI);
    const float fv = sigmoid_fast(acc_f[r] + bF);
    const float ov = sigmoid_fast(acc_o[r] + bO);
    const float gv = tanh_fast(acc_g[r] + bG);
    const float cn = fv * Cst[idx] + iv * gv;
    Cst[idx]  = cn;
    Hout[idx] = (_Float16)(ov * tanh_fast(cn));
  }
}

// ---- Final FC: out = h @ fc_w^T + fc_b  (128 x 1000, N padded to 1008) ----
__global__ __launch_bounds__(128) void fc_wmma(
    const _Float16* __restrict__ A,     // h   [BB][HD] f16
    const _Float16* __restrict__ W,     // fcW [NCP][HD] f16 (zero-padded rows)
    const float* __restrict__ bias,     // [NC]
    float* __restrict__ out)            // [BB][NC] f32
{
  const int lane = threadIdx.x & 31;
  const int gw   = blockIdx.x * 4 + (threadIdx.x >> 5);   // 504 waves
  const int m0   = (gw / 63) * 16;
  const int n0   = (gw % 63) * 16;
  const int half = lane >> 4;
  const int nl   = lane & 15;

  const _Float16* wrow = W + (size_t)(n0 + nl) * HD;
  const _Float16* arow = A + (size_t)(m0 + nl) * HD;

  v8f acc = {};
  v16h a = load_a_frag(arow, half);
  v16h b = load_b_frag(wrow, 0, half);
  for (int kk = 0; kk < HD; kk += 32) {
    const int kn = (kk + 32 < HD) ? (kk + 32) : 0;
    const v16h a2 = load_a_frag(arow + kn, half);
    const v16h b2 = load_b_frag(wrow, kn, half);
    acc = wmma_f16(a, b, acc);
    a = a2; b = b2;
  }
  const int n = n0 + nl;
  if (n < NC) {                         // divergence only after all WMMAs
    const float bv = bias[n];
    #pragma unroll
    for (int r = 0; r < 8; ++r)
      out[(size_t)(m0 + r + 8 * half) * NC + n] = acc[r] + bv;
  }
}

// ---- One-time prep kernels ----

// center tap of [N][K][3][3] conv weight -> f16 [N][K]
__global__ void center_tap_f16(const float* __restrict__ w, _Float16* __restrict__ o, int n) {
  int i = blockIdx.x * blockDim.x + threadIdx.x;
  if (i < n) o[i] = (_Float16)w[(size_t)i * 9 + 4];
}

// fc_w [1000][512] f32 -> [1008][512] f16 with zero pad rows
__global__ void fc_pad_f16(const float* __restrict__ w, _Float16* __restrict__ o) {
  int i = blockIdx.x * blockDim.x + threadIdx.x;
  if (i < NCP * HD) {
    int nn = i >> 9;   // /512
    o[i] = (nn < NC) ? (_Float16)w[i] : (_Float16)0.0f;
  }
}

// x [B][T][256] f32 -> xh [T][B][256] f16 (time-major for per-step locality)
__global__ void x_to_f16_tbc(const float* __restrict__ x, _Float16* __restrict__ xh) {
  int i = blockIdx.x * blockDim.x + threadIdx.x;
  if (i < TT * BB * IND) {
    int k   = i & (IND - 1);
    int rem = i >> 8;
    int b   = rem & (BB - 1);
    int t   = rem >> 7;
    xh[i] = (_Float16)x[((size_t)b * TT + t) * IND + k];
  }
}

__global__ void init_state(_Float16* __restrict__ hh, float* __restrict__ c) {
  int i = blockIdx.x * blockDim.x + threadIdx.x;
  if (i < BB * HD) { hh[i] = (_Float16)0.0f; c[i] = 0.0f; }
}

extern "C" void kernel_launch(void* const* d_in, const int* in_sizes, int n_in,
                              void* d_out, int out_size, void* d_ws, size_t ws_size,
                              hipStream_t stream) {
  const float* x    = (const float*)d_in[0];
  const float* w0   = (const float*)d_in[1];
  const float* b0   = (const float*)d_in[2];
  const float* w1   = (const float*)d_in[3];
  const float* b1   = (const float*)d_in[4];
  const float* fc_w = (const float*)d_in[5];
  const float* fc_b = (const float*)d_in[6];
  float* out = (float*)d_out;
  (void)in_sizes; (void)n_in; (void)out_size; (void)ws_size;

  // workspace carve-out (256B aligned slices; total ~25 MB)
  char* ws = (char*)d_ws;
  size_t off = 0;
  auto carve = [&](size_t bytes) {
    char* p = ws + off;
    off = (off + bytes + 255) & ~(size_t)255;
    return p;
  };
  _Float16* W0h  = (_Float16*)carve((size_t)4 * HD * (IND + HD) * 2); // 2048x768  f16
  _Float16* W1h  = (_Float16*)carve((size_t)4 * HD * (2 * HD)   * 2); // 2048x1024 f16
  _Float16* FCh  = (_Float16*)carve((size_t)NCP * HD * 2);            // 1008x512  f16
  _Float16* xh   = (_Float16*)carve((size_t)TT * BB * IND * 2);       // T,B,256   f16
  _Float16* hh   = (_Float16*)carve((size_t)BB * HD * 2);             // h         f16
  _Float16* hmid = (_Float16*)carve((size_t)BB * HD * 2);             // inter-cell h
  float*    cst  = (float*)   carve((size_t)BB * HD * 4);             // cell state f32

  // one-time conversions (weights stay hot in 192MB L2 for all 256 steps)
  const int n0e = 4 * HD * (IND + HD);
  center_tap_f16<<<(n0e + 255) / 256, 256, 0, stream>>>(w0, W0h, n0e);
  const int n1e = 4 * HD * (2 * HD);
  center_tap_f16<<<(n1e + 255) / 256, 256, 0, stream>>>(w1, W1h, n1e);
  fc_pad_f16<<<(NCP * HD + 255) / 256, 256, 0, stream>>>(fc_w, FCh);
  x_to_f16_tbc<<<(TT * BB * IND + 255) / 256, 256, 0, stream>>>(x, xh);
  init_state<<<(BB * HD + 255) / 256, 256, 0, stream>>>(hh, cst);

  // sequential scan: 2 fused cell kernels per step
  for (int t = 0; t < TT; ++t) {
    // cell0: combined = [x_t | h] , K = 768
    lstm_cell_wmma<<<64, 128, 0, stream>>>(xh + (size_t)t * BB * IND, IND, IND,
                                           hh, W0h, IND + HD, b0, cst, hmid);
    // cell1: combined = [h_mid | h_mid] , K = 1024 (both halves alias hmid)
    lstm_cell_wmma<<<64, 128, 0, stream>>>(hmid, HD, HD,
                                           hmid, W1h, 2 * HD, b1, cst, hh);
  }

  // classifier
  fc_wmma<<<126, 128, 0, stream>>>(hh, FCh, fc_b, out);
}
